// SDPAMultiHeadSelfAttention_10316511445115
// MI455X (gfx1250) — matrix operand order repaired and checked by reading, not compiled, not run on hardware
//
#include <hip/hip_runtime.h>
#include <hip/hip_bf16.h>

typedef __bf16 bf16_t;
typedef __attribute__((ext_vector_type(8)))  __bf16 v8bf;
typedef __attribute__((ext_vector_type(16))) __bf16 v16bf;
typedef __attribute__((ext_vector_type(8)))  float  v8f;

#define BATCH   2
#define SEQ     2048
#define EMBED   1024
#define NHEADS  16
#define HDIM    64
#define NEG_INF (-1000000000.0f)

// ---------------------------------------------------------------------------
// Fragment helpers (wave32, WMMA fragments in contiguous k order, k = 16*(lane/16) + elem, in A and B)
// ---------------------------------------------------------------------------
static __device__ inline v16bf cat8(v8bf lo, v8bf hi) {
  return __builtin_shufflevector(lo, hi, 0,1,2,3,4,5,6,7,8,9,10,11,12,13,14,15);
}

// A-matrix 16x32 bf16: lane holds row m = lane%16.
// lanes 0-15: elems 0..15 = K0..15
// lanes 16-31: elems 0..15 = K16..31
static __device__ inline v16bf load_a_frag(const bf16_t* __restrict__ A, int lda,
                                           int m0, int k0) {
  int lane = threadIdx.x & 31;
  int r = lane & 15, hi = lane >> 4;
  const bf16_t* base = A + (size_t)(m0 + r) * lda + k0 + hi * 16;
  v8bf lo = *(const v8bf*)(base);
  v8bf hh = *(const v8bf*)(base + 8);
  return cat8(lo, hh);
}

// B-matrix 32x16 bf16 for C = A * W^T with W row-major [N,K]:
// lane holds column n = n0 + lane%16; elems 0..15 = K (k0 + (lane>=16?16:0) + e)
static __device__ inline v16bf load_b_frag(const bf16_t* __restrict__ W, int ldb,
                                           int n0, int k0) {
  int lane = threadIdx.x & 31;
  return *(const v16bf*)(W + (size_t)(n0 + (lane & 15)) * ldb + k0 + (lane >> 4) * 16);
}

static __device__ inline float rowmax16(float v) {
  #pragma unroll
  for (int o = 1; o < 16; o <<= 1) v = fmaxf(v, __shfl_xor(v, o, 32));
  return v;
}
static __device__ inline float rowsum16(float v) {
  #pragma unroll
  for (int o = 1; o < 16; o <<= 1) v += __shfl_xor(v, o, 32);
  return v;
}

// ---------------------------------------------------------------------------
// fp32 -> bf16 conversion
// ---------------------------------------------------------------------------
__global__ __launch_bounds__(256)
void k_f32_to_bf16(bf16_t* __restrict__ dst, const float* __restrict__ src, int n) {
  int i = blockIdx.x * blockDim.x + threadIdx.x;
  if (i < n) dst[i] = (bf16_t)src[i];
}

// ---------------------------------------------------------------------------
// GEMM: C[M,N] = A[M,K] * W[N,K]^T, bf16 inputs, fp32 out.
// block = 128 threads (4 waves); wave tile 32x32 (2 A-frag x 2 B-frag ->
// 4 WMMAs, 1.0 fragment-loads per WMMA); block tile 32x128.
// grid = (M/32, N/128)
// ---------------------------------------------------------------------------
__global__ __launch_bounds__(128)
void k_gemm_bf16_wmma(float* __restrict__ C, const bf16_t* __restrict__ A,
                      const bf16_t* __restrict__ W, int N, int K) {
  const int lane = threadIdx.x & 31;
  const int wave = threadIdx.x >> 5;
  const int m0 = blockIdx.x * 32;
  const int n0 = blockIdx.y * 128 + wave * 32;
  v8f c00 = {}, c01 = {}, c10 = {}, c11 = {};
  for (int k0 = 0; k0 < K; k0 += 32) {
    v16bf a0 = load_a_frag(A, K, m0, k0);
    v16bf a1 = load_a_frag(A, K, m0 + 16, k0);
    v16bf b0 = load_b_frag(W, K, n0, k0);
    v16bf b1 = load_b_frag(W, K, n0 + 16, k0);
    c00 = __builtin_amdgcn_wmma_f32_16x16x32_bf16(false, a0, false, b0, (short)0, c00, false, false);
    c01 = __builtin_amdgcn_wmma_f32_16x16x32_bf16(false, a0, false, b1, (short)0, c01, false, false);
    c10 = __builtin_amdgcn_wmma_f32_16x16x32_bf16(false, a1, false, b0, (short)0, c10, false, false);
    c11 = __builtin_amdgcn_wmma_f32_16x16x32_bf16(false, a1, false, b1, (short)0, c11, false, false);
  }
  const int hi = lane >> 4, nl = lane & 15;
  #pragma unroll
  for (int i = 0; i < 8; ++i) {
    size_t r0 = (size_t)(m0 + hi * 8 + i);
    size_t r1 = (size_t)(m0 + 16 + hi * 8 + i);
    C[r0 * N + n0 + nl]      = c00[i];
    C[r0 * N + n0 + 16 + nl] = c01[i];
    C[r1 * N + n0 + nl]      = c10[i];
    C[r1 * N + n0 + 16 + nl] = c11[i];
  }
}

// ---------------------------------------------------------------------------
// RoPE + pack: qkv fp32 [B,S,3E] -> Qb/Kb bf16 [B,H,S,D] (Q pre-scaled 1/8),
// Vt bf16 [B,H,D,S] (transposed so PV B-fragments are contiguous).
// one thread per (b,h,s,d); total B*H*S*D threads.
// ---------------------------------------------------------------------------
__global__ __launch_bounds__(256)
void k_rope_pack(bf16_t* __restrict__ Qb, bf16_t* __restrict__ Kb,
                 bf16_t* __restrict__ Vt, const float* __restrict__ qkv) {
  int t = blockIdx.x * blockDim.x + threadIdx.x;
  int d = t & (HDIM - 1);
  int s = (t >> 6) & (SEQ - 1);
  int h = (t >> 17) & (NHEADS - 1);
  int b = t >> 21;
  size_t rowbase = (size_t)(b * SEQ + s) * (3 * EMBED);
  size_t he = (size_t)h * HDIM;
  float q = qkv[rowbase + he + d];
  float k = qkv[rowbase + EMBED + he + d];
  float v = qkv[rowbase + 2 * EMBED + he + d];
  int i = d & 31, half = d >> 5;
  int d2 = half ? (d - 32) : (d + 32);
  float qp = qkv[rowbase + he + d2];
  float kp = qkv[rowbase + EMBED + he + d2];
  // inv_freq[i] = 10000^(-i/32) = exp(-i * ln(10000)/32)
  float theta = (float)s * __expf(-(float)i * (9.210340371976184f / 32.0f));
  float cs = __cosf(theta), sn = __sinf(theta);
  float sgn = half ? 1.0f : -1.0f;
  float qr = q * cs + sgn * qp * sn;
  float kr = k * cs + sgn * kp * sn;
  size_t bh = (size_t)b * NHEADS + h;
  Qb[(bh * SEQ + s) * HDIM + d] = (bf16_t)(qr * 0.125f);  // fold 1/sqrt(64)
  Kb[(bh * SEQ + s) * HDIM + d] = (bf16_t)kr;
  Vt[(bh * HDIM + d) * SEQ + s] = (bf16_t)v;
}

// ---------------------------------------------------------------------------
// Flash attention: one wave per (b, h, 16-query tile).
// Q kept in registers (two A-frags). 32 keys per iteration:
//   4 WMMAs for QK^T, online softmax (shfl_xor row reductions),
//   P relaid C->A via 1KB LDS, 4 WMMAs for P*V.
// Output written bf16 to attn[B,S,E] for the out-projection GEMM.
// ---------------------------------------------------------------------------
__global__ __launch_bounds__(32)
void k_flash_attn(bf16_t* __restrict__ attn, const bf16_t* __restrict__ Qb,
                  const bf16_t* __restrict__ Kb, const bf16_t* __restrict__ Vt,
                  const unsigned char* __restrict__ kpm) {
  __shared__ bf16_t pTile[16 * 32];
  const int lane = threadIdx.x & 31;
  const int nl = lane & 15, hi = lane >> 4;
  const int q0 = blockIdx.x * 16;
  const int h  = blockIdx.y;
  const int b  = blockIdx.z;
  const size_t bh = (size_t)b * NHEADS + h;
  const bf16_t* Qbase = Qb + bh * SEQ * HDIM;
  const bf16_t* Kbase = Kb + bh * SEQ * HDIM;
  const bf16_t* Vbase = Vt + bh * HDIM * SEQ;

  // Q A-frags: dims 0..31 and 32..63, row = q0 + lane%16
  const bf16_t* qrow = Qbase + (size_t)(q0 + nl) * HDIM;
  v16bf qa0 = cat8(*(const v8bf*)(qrow + hi * 16),      *(const v8bf*)(qrow + 8 + hi * 16));
  v16bf qa1 = cat8(*(const v8bf*)(qrow + 32 + hi * 16), *(const v8bf*)(qrow + 40 + hi * 16));

  v8f acc0 = {}, acc1 = {}, acc2 = {}, acc3 = {};
  float mrow[8], lrow[8];
  #pragma unroll
  for (int i = 0; i < 8; ++i) { mrow[i] = -__builtin_inff(); lrow[i] = 0.0f; }

  for (int kt = 0; kt < SEQ; kt += 32) {
    // ---- scores: two 16-key C tiles, K-dim (=64) in two 32-chunks ----
    v8f s0 = {}, s1 = {};
    {
      const bf16_t* k0row = Kbase + (size_t)(kt + nl) * HDIM + hi * 16;
      v16bf kb00 = *(const v16bf*)(k0row);       // dims 0..31
      v16bf kb01 = *(const v16bf*)(k0row + 32);  // dims 32..63
      s0 = __builtin_amdgcn_wmma_f32_16x16x32_bf16(false, qa0, false, kb00, (short)0, s0, false, false);
      s0 = __builtin_amdgcn_wmma_f32_16x16x32_bf16(false, qa1, false, kb01, (short)0, s0, false, false);
      const bf16_t* k1row = Kbase + (size_t)(kt + 16 + nl) * HDIM + hi * 16;
      v16bf kb10 = *(const v16bf*)(k1row);
      v16bf kb11 = *(const v16bf*)(k1row + 32);
      s1 = __builtin_amdgcn_wmma_f32_16x16x32_bf16(false, qa0, false, kb10, (short)0, s1, false, false);
      s1 = __builtin_amdgcn_wmma_f32_16x16x32_bf16(false, qa1, false, kb11, (short)0, s1, false, false);
    }
    // ---- key padding mask (column = key index held by this lane) ----
    const bool msk0 = kpm[(size_t)b * SEQ + kt + nl] != 0;
    const bool msk1 = kpm[(size_t)b * SEQ + kt + 16 + nl] != 0;
    #pragma unroll
    for (int i = 0; i < 8; ++i) {
      if (msk0) s0[i] = NEG_INF;
      if (msk1) s1[i] = NEG_INF;
    }
    // ---- online softmax (row i of this lane = hi*8 + i) ----
    float scl[8];
    #pragma unroll
    for (int i = 0; i < 8; ++i) {
      float mx = rowmax16(fmaxf(s0[i], s1[i]));
      float mnew = fmaxf(mrow[i], mx);
      float sc = __expf(mrow[i] - mnew);
      float p0 = __expf(s0[i] - mnew);
      float p1 = __expf(s1[i] - mnew);
      lrow[i] = lrow[i] * sc + rowsum16(p0 + p1);
      mrow[i] = mnew;
      scl[i] = sc;
      s0[i] = p0; s1[i] = p1;
    }
    #pragma unroll
    for (int i = 0; i < 8; ++i) {
      acc0[i] *= scl[i]; acc1[i] *= scl[i]; acc2[i] *= scl[i]; acc3[i] *= scl[i];
    }
    // ---- P: C layout -> LDS (16 rows x 32 keys) -> A-frag ----
    __syncthreads();  // WAR vs previous iteration's LDS reads
    #pragma unroll
    for (int i = 0; i < 8; ++i) {
      pTile[(hi * 8 + i) * 32 + nl]      = (bf16_t)s0[i];
      pTile[(hi * 8 + i) * 32 + 16 + nl] = (bf16_t)s1[i];
    }
    __syncthreads();
    const bf16_t* prow = &pTile[nl * 32];
    v16bf pa = cat8(*(const v8bf*)(prow + hi * 16), *(const v8bf*)(prow + 8 + hi * 16));
    // ---- P(16x32keys) * V(32keys x 64d): 4 WMMAs over 16-wide d blocks ----
    {
      const bf16_t* vb = Vbase + (size_t)nl * SEQ + kt + hi * 16;
      v16bf v0 = *(const v16bf*)(vb + (size_t)(0 * 16) * SEQ);
      v16bf v1 = *(const v16bf*)(vb + (size_t)(1 * 16) * SEQ);
      v16bf v2 = *(const v16bf*)(vb + (size_t)(2 * 16) * SEQ);
      v16bf v3 = *(const v16bf*)(vb + (size_t)(3 * 16) * SEQ);
      acc0 = __builtin_amdgcn_wmma_f32_16x16x32_bf16(false, pa, false, v0, (short)0, acc0, false, false);
      acc1 = __builtin_amdgcn_wmma_f32_16x16x32_bf16(false, pa, false, v1, (short)0, acc1, false, false);
      acc2 = __builtin_amdgcn_wmma_f32_16x16x32_bf16(false, pa, false, v2, (short)0, acc2, false, false);
      acc3 = __builtin_amdgcn_wmma_f32_16x16x32_bf16(false, pa, false, v3, (short)0, acc3, false, false);
    }
  }

  // ---- normalize and store bf16 to attn[B,S,E], e = h*64 + d ----
  #pragma unroll
  for (int i = 0; i < 8; ++i) {
    float inv = 1.0f / lrow[i];
    int s = q0 + hi * 8 + i;
    size_t base = (size_t)(b * SEQ + s) * EMBED + (size_t)h * HDIM;
    attn[base + 0 * 16 + nl] = (bf16_t)(acc0[i] * inv);
    attn[base + 1 * 16 + nl] = (bf16_t)(acc1[i] * inv);
    attn[base + 2 * 16 + nl] = (bf16_t)(acc2[i] * inv);
    attn[base + 3 * 16 + nl] = (bf16_t)(acc3[i] * inv);
  }
}

// ---------------------------------------------------------------------------
// Host-side launch
// ---------------------------------------------------------------------------
extern "C" void kernel_launch(void* const* d_in, const int* in_sizes, int n_in,
                              void* d_out, int out_size, void* d_ws, size_t ws_size,
                              hipStream_t stream) {
  (void)in_sizes; (void)n_in; (void)out_size; (void)ws_size;
  const float* x            = (const float*)d_in[0];          // [B,S,E]
  const unsigned char* kpm  = (const unsigned char*)d_in[1];  // [B,S] bool
  const float* Wqkv         = (const float*)d_in[2];          // [3E,E]
  const float* Wout         = (const float*)d_in[3];          // [E,E]
  float* out                = (float*)d_out;                  // [B,S,E]

  const int M  = BATCH * SEQ;          // 4096
  const int nX = M * EMBED;            // 4,194,304
  const int nWq = 3 * EMBED * EMBED;   // 3,145,728
  const int nWo = EMBED * EMBED;       // 1,048,576

  char* ws = (char*)d_ws;
  size_t off = 0;
  auto carve = [&](size_t bytes) -> char* {
    char* p = ws + off;
    off += (bytes + 255) & ~(size_t)255;
    return p;
  };
  bf16_t* xb    = (bf16_t*)carve((size_t)nX  * 2);
  bf16_t* Wqkvb = (bf16_t*)carve((size_t)nWq * 2);
  bf16_t* Woutb = (bf16_t*)carve((size_t)nWo * 2);
  float*  qkv   = (float*) carve((size_t)M * 3 * EMBED * 4);
  bf16_t* Qb    = (bf16_t*)carve((size_t)M * EMBED * 2);
  bf16_t* Kb    = (bf16_t*)carve((size_t)M * EMBED * 2);
  bf16_t* Vt    = (bf16_t*)carve((size_t)M * EMBED * 2);
  bf16_t* attn  = (bf16_t*)carve((size_t)M * EMBED * 2);

  // 1) downconvert inputs/weights to bf16
  k_f32_to_bf16<<<(nX  + 255) / 256, 256, 0, stream>>>(xb,    x,    nX);
  k_f32_to_bf16<<<(nWq + 255) / 256, 256, 0, stream>>>(Wqkvb, Wqkv, nWq);
  k_f32_to_bf16<<<(nWo + 255) / 256, 256, 0, stream>>>(Woutb, Wout, nWo);

  // 2) qkv = x * Wqkv^T   (M=4096, N=3072, K=1024)
  k_gemm_bf16_wmma<<<dim3(M / 32, (3 * EMBED) / 128), 128, 0, stream>>>(
      qkv, xb, Wqkvb, 3 * EMBED, EMBED);

  // 3) RoPE + pack into per-head layouts
  k_rope_pack<<<(BATCH * NHEADS * SEQ * HDIM) / 256, 256, 0, stream>>>(Qb, Kb, Vt, qkv);

  // 4) flash attention
  k_flash_attn<<<dim3(SEQ / 16, NHEADS, BATCH), 32, 0, stream>>>(attn, Qb, Kb, Vt, kpm);

  // 5) out = attn * Wout^T  (M=4096, N=1024, K=1024)
  k_gemm_bf16_wmma<<<dim3(M / 32, EMBED / 128), 128, 0, stream>>>(
      out, attn, Woutb, EMBED, EMBED);
}
